// CaptioningRNN_27255862460474
// MI455X (gfx1250) — compile-verified, matrix operand order checked by reading, not run on hardware
//
#include <hip/hip_runtime.h>
#include <hip/hip_bf16.h>
#include <string.h>

// ---------- types ----------
typedef __bf16 v16bf __attribute__((ext_vector_type(16)));
typedef float  v8f   __attribute__((ext_vector_type(8)));
typedef unsigned int u32x4 __attribute__((ext_vector_type(4)));

union FragBF { u32x4 u[2]; v16bf v; };
union Pack8  { u32x4 u;    __bf16 h[8]; };

// Problem constants
#define NB   128      // batch
#define TT   32       // timesteps (T-1)
#define DW   512      // embed width == Dfeat
#define HH   1024     // hidden
#define VV   16384    // vocab
#define NT   (NB*TT)  // 4096 rows for x / h / scores

// ---------- async global->LDS (gfx1250 ASYNCcnt path), with safe fallback ----
#if __has_builtin(__builtin_amdgcn_global_load_async_to_lds_b128)
#define HAVE_ASYNC_LDS 1
// Builtin signature (from hipcc diagnostic): (v4i AS(1)*, v4i AS(3)*, imm, imm)
typedef int v4i_t __attribute__((vector_size(16)));
typedef __attribute__((address_space(1))) v4i_t as1_v4i;
typedef __attribute__((address_space(3))) v4i_t as3_v4i;
#endif

__device__ __forceinline__ void copy16_to_lds(const void* g, void* l) {
#ifdef HAVE_ASYNC_LDS
  __builtin_amdgcn_global_load_async_to_lds_b128((as1_v4i*)g, (as3_v4i*)l, 0, 0);
#else
  *(u32x4*)l = *(const u32x4*)g;
#endif
}
__device__ __forceinline__ void lds_copy_drain() {
#ifdef HAVE_ASYNC_LDS
#if __has_builtin(__builtin_amdgcn_s_wait_asynccnt)
  __builtin_amdgcn_s_wait_asynccnt(0);
#else
  asm volatile("s_wait_asynccnt 0" ::: "memory");
#endif
#endif
}

// ---------- tiny utility kernels ----------
__global__ void k_zero_out(float* out) { out[0] = 0.0f; }

__global__ void k_f32_to_bf16(const float* __restrict__ src, __bf16* __restrict__ dst, int n) {
  for (int i = blockIdx.x * blockDim.x + threadIdx.x; i < n; i += gridDim.x * blockDim.x)
    dst[i] = (__bf16)src[i];
}

// W_vocab (H x V, f32) -> Wvt (V x H, bf16), tiled transpose
__global__ void k_transpose_convert(const float* __restrict__ src, __bf16* __restrict__ dst) {
  __shared__ float tile[32][33];
  int v0 = blockIdx.x * 32, h0 = blockIdx.y * 32;
  tile[threadIdx.y][threadIdx.x] = src[(h0 + threadIdx.y) * VV + v0 + threadIdx.x];
  __syncthreads();
  dst[(size_t)(v0 + threadIdx.y) * HH + h0 + threadIdx.x] = (__bf16)tile[threadIdx.x][threadIdx.y];
}

// x[t*128+n][w] = bf16(W_embed[captions[n][t]][w]),  t in 0..31
__global__ void k_embed_gather(const float* __restrict__ wembed, const int* __restrict__ captions,
                               __bf16* __restrict__ xbf) {
  int idx = blockIdx.x * blockDim.x + threadIdx.x;   // over NT*DW
  if (idx >= NT * DW) return;
  int r = idx / DW, w = idx % DW;
  int n = r % NB, t = r / NB;
  int cap = captions[n * 33 + t];
  xbf[idx] = (__bf16)wembed[(size_t)cap * DW + w];
}

// ---------- generic 64x64 bf16 WMMA GEMM ----------
// C[M,N] = A[M,K] @ B[K,N]  (+bias[N]) (+addend[M,ldadd]) (tanh?) -> f32 and/or bf16
#define APAD 40  // padded LDS K-stride (multiple of 8; 20 dwords -> conflict-free rows)
__global__ void __launch_bounds__(256)
k_gemm64(const __bf16* __restrict__ A, int lda,
         const __bf16* __restrict__ B, int ldb,
         const float* __restrict__ bias,
         const float* __restrict__ addend, int ldadd,
         float* __restrict__ outf, __bf16* __restrict__ outbf, int ldc,
         int K, int do_tanh) {
  __shared__ __bf16 As[64 * APAD];
  __shared__ __bf16 Bs[64 * APAD];

  const int tid  = threadIdx.x;
  const int lane = tid & 31, wid = tid >> 5;
  const int l16  = lane & 15, half = lane >> 4;
  const int wm   = wid & 3, wn = wid >> 2;       // wave -> (Mtile, N pair)
  const int m0   = blockIdx.x * 64, n0 = blockIdx.y * 64;

  v8f acc0 = {0.f,0.f,0.f,0.f,0.f,0.f,0.f,0.f};
  v8f acc1 = {0.f,0.f,0.f,0.f,0.f,0.f,0.f,0.f};

  for (int kb = 0; kb < K; kb += 32) {
    { // stage A: 64 rows x 32 halves (async direct-to-LDS when available)
      int row = tid >> 2, seg = (tid & 3) * 8;
      copy16_to_lds(&A[(size_t)(m0 + row) * lda + kb + seg], &As[row * APAD + seg]);
    }
    { // stage B transposed: Bs[n][k] (overlaps with in-flight async A transfer)
      int k = tid >> 3, nseg = (tid & 7) * 8;
      Pack8 bv; bv.u = *(const u32x4*)&B[(size_t)(kb + k) * ldb + n0 + nseg];
      #pragma unroll
      for (int q = 0; q < 8; ++q) Bs[(nseg + q) * APAD + k] = bv.h[q];
    }
    lds_copy_drain();
    __syncthreads();

    FragBF fa;
    int rowA = wm * 16 + l16;
    fa.u[0] = *(u32x4*)&As[rowA * APAD + 8 * half];
    fa.u[1] = *(u32x4*)&As[rowA * APAD + 16 + 8 * half];

    FragBF fb0, fb1;
    int colB0 = (2 * wn) * 16 + l16, colB1 = (2 * wn + 1) * 16 + l16;
    fb0.u[0] = *(u32x4*)&Bs[colB0 * APAD + 8 * half];
    fb0.u[1] = *(u32x4*)&Bs[colB0 * APAD + 16 + 8 * half];
    fb1.u[0] = *(u32x4*)&Bs[colB1 * APAD + 8 * half];
    fb1.u[1] = *(u32x4*)&Bs[colB1 * APAD + 16 + 8 * half];

    acc0 = __builtin_amdgcn_wmma_f32_16x16x32_bf16(false, fa.v, false, fb0.v,
                                                   (short)0, acc0, false, false);
    acc1 = __builtin_amdgcn_wmma_f32_16x16x32_bf16(false, fa.v, false, fb1.v,
                                                   (short)0, acc1, false, false);
    __syncthreads();
  }

  #pragma unroll
  for (int s = 0; s < 2; ++s) {
    v8f acc = s ? acc1 : acc0;
    int colg = n0 + (2 * wn + s) * 16 + l16;
    #pragma unroll
    for (int i = 0; i < 8; ++i) {
      int rowg = m0 + wm * 16 + i + 8 * half;
      float v = acc[i];
      if (bias)   v += bias[colg];
      if (addend) v += addend[(size_t)rowg * ldadd + colg];
      if (do_tanh) v = tanhf(v);
      if (outf)  outf[(size_t)rowg * ldc + colg] = v;
      if (outbf) outbf[(size_t)rowg * ldc + colg] = (__bf16)v;
    }
  }
}

// ---------- fused vocab GEMM + log-softmax NLL ----------
// Each block owns a 16-row tile of the 4096 (t,n) rows; 8 waves sweep V in
// groups of 4 N-tiles (64 columns) so one A fragment feeds 4 independent WMMAs.
#define HPAD 1048  // padded LDS stride for 16x1024 A tile (524 dwords -> conflict-free)
__global__ void __launch_bounds__(256)
k_vocab_loss(const __bf16* __restrict__ hall,   // [NT][HH] bf16, row r = t*128+n
             const __bf16* __restrict__ wvt,    // [VV][HH] bf16 (transposed W_vocab)
             const float* __restrict__ bvocab,  // [VV]
             const int*  __restrict__ captions, // [128][33]
             float* __restrict__ out) {
  __shared__ __bf16 Alds[16 * HPAD];
  __shared__ float red_m[8][16];
  __shared__ float red_s[8][16];
  __shared__ float tgt_score_s[16];
  __shared__ int   tgt_col_s[16];

  const int tid  = threadIdx.x;
  const int lane = tid & 31, wid = tid >> 5;
  const int l16  = lane & 15, half = lane >> 4;
  const int r0   = blockIdx.x * 16;

  { // stage 16 rows x 1024 halves of h (async direct-to-LDS when available)
    int row = tid >> 4, s0 = tid & 15;
    const u32x4* src = (const u32x4*)&hall[(size_t)(r0 + row) * HH];
    u32x4* dst = (u32x4*)&Alds[row * HPAD];
    #pragma unroll
    for (int i = 0; i < 8; ++i) copy16_to_lds(&src[s0 + i * 16], &dst[s0 + i * 16]);
  }
  if (tid < 16) {
    int g = r0 + tid, t = g / NB, n = g % NB;
    tgt_col_s[tid]   = captions[n * 33 + t + 1];
    tgt_score_s[tid] = 0.0f;
  }
  lds_copy_drain();
  __syncthreads();

  float rm[8], rs[8];
  #pragma unroll
  for (int i = 0; i < 8; ++i) { rm[i] = -__builtin_inff(); rs[i] = 0.0f; }

  const __bf16* aldsbase = &Alds[l16 * HPAD];

  for (int g = wid; g < VV / 64; g += 8) {     // 256 groups of 4 N-tiles
    int colb = g * 64 + l16;
    const __bf16* br0 = &wvt[(size_t)(colb)      * HH];
    const __bf16* br1 = &wvt[(size_t)(colb + 16) * HH];
    const __bf16* br2 = &wvt[(size_t)(colb + 32) * HH];
    const __bf16* br3 = &wvt[(size_t)(colb + 48) * HH];
    // prefetch this wave's next group of B rows (L2-resident W_vocab)
    if (colb + 512 < VV) __builtin_prefetch(&wvt[(size_t)(colb + 512) * HH], 0, 0);

    v8f acc[4];
    #pragma unroll
    for (int s = 0; s < 4; ++s) acc[s] = (v8f){0.f,0.f,0.f,0.f,0.f,0.f,0.f,0.f};

    #pragma unroll 4
    for (int kc = 0; kc < 32; ++kc) {
      int kb = kc * 32;
      FragBF fa, fb0, fb1, fb2, fb3;
      fa.u[0]  = *(const u32x4*)&aldsbase[kb + 8 * half];
      fa.u[1]  = *(const u32x4*)&aldsbase[kb + 16 + 8 * half];
      fb0.u[0] = *(const u32x4*)&br0[kb + 8 * half];
      fb0.u[1] = *(const u32x4*)&br0[kb + 16 + 8 * half];
      fb1.u[0] = *(const u32x4*)&br1[kb + 8 * half];
      fb1.u[1] = *(const u32x4*)&br1[kb + 16 + 8 * half];
      fb2.u[0] = *(const u32x4*)&br2[kb + 8 * half];
      fb2.u[1] = *(const u32x4*)&br2[kb + 16 + 8 * half];
      fb3.u[0] = *(const u32x4*)&br3[kb + 8 * half];
      fb3.u[1] = *(const u32x4*)&br3[kb + 16 + 8 * half];
      acc[0] = __builtin_amdgcn_wmma_f32_16x16x32_bf16(false, fa.v, false, fb0.v,
                                                       (short)0, acc[0], false, false);
      acc[1] = __builtin_amdgcn_wmma_f32_16x16x32_bf16(false, fa.v, false, fb1.v,
                                                       (short)0, acc[1], false, false);
      acc[2] = __builtin_amdgcn_wmma_f32_16x16x32_bf16(false, fa.v, false, fb2.v,
                                                       (short)0, acc[2], false, false);
      acc[3] = __builtin_amdgcn_wmma_f32_16x16x32_bf16(false, fa.v, false, fb3.v,
                                                       (short)0, acc[3], false, false);
    }

    #pragma unroll
    for (int s = 0; s < 4; ++s) {
      int col = colb + s * 16;
      float bias = bvocab[col];
      #pragma unroll
      for (int i = 0; i < 8; ++i) {
        float sc = acc[s][i] + bias;
        int rowloc = i + 8 * half;
        if (col == tgt_col_s[rowloc]) tgt_score_s[rowloc] = sc;
        float mo = rm[i], mn = fmaxf(mo, sc);
        rs[i] = rs[i] * __expf(mo - mn) + __expf(sc - mn);
        rm[i] = mn;
      }
    }
  }

  // reduce (m,s) across the 16 lanes of each half
  #pragma unroll
  for (int off = 1; off < 16; off <<= 1) {
    #pragma unroll
    for (int i = 0; i < 8; ++i) {
      float om = __shfl_xor(rm[i], off, 32);
      float os = __shfl_xor(rs[i], off, 32);
      float mn = fmaxf(rm[i], om);
      rs[i] = rs[i] * __expf(rm[i] - mn) + os * __expf(om - mn);
      rm[i] = mn;
    }
  }
  if (l16 == 0) {
    #pragma unroll
    for (int i = 0; i < 8; ++i) {
      red_m[wid][half * 8 + i] = rm[i];
      red_s[wid][half * 8 + i] = rs[i];
    }
  }
  __syncthreads();

  if (tid < 16) {
    float m = -__builtin_inff(), s = 0.0f;
    #pragma unroll
    for (int w = 0; w < 8; ++w) {
      float om = red_m[w][tid], os = red_s[w][tid];
      float mn = fmaxf(m, om);
      s = s * __expf(m - mn) + os * __expf(om - mn);
      m = mn;
    }
    float lse = __logf(s) + m;
    float nll = (tgt_col_s[tid] != 0) ? (lse - tgt_score_s[tid]) : 0.0f;
    atomicAdd(out, nll * (1.0f / (float)NB));
  }
}

// ---------- launch ----------
extern "C" void kernel_launch(void* const* d_in, const int* in_sizes, int n_in,
                              void* d_out, int out_size, void* d_ws, size_t ws_size,
                              hipStream_t stream) {
  const float* features = (const float*)d_in[0];   // 128x512
  const int*   captions = (const int*)d_in[1];     // 128x33
  const float* W_proj   = (const float*)d_in[2];   // 512x1024
  const float* b_proj   = (const float*)d_in[3];   // 1024
  const float* W_embed  = (const float*)d_in[4];   // 16384x512
  const float* Wx       = (const float*)d_in[5];   // 512x1024
  const float* Wh       = (const float*)d_in[6];   // 1024x1024
  const float* bvec     = (const float*)d_in[7];   // 1024
  const float* W_vocab  = (const float*)d_in[8];   // 1024x16384
  const float* b_vocab  = (const float*)d_in[9];   // 16384
  float* out = (float*)d_out;

  char* w = (char*)d_ws;
  __bf16* wprojbf = (__bf16*)(w);                         // 1,048,576 B
  __bf16* wxbf    = (__bf16*)(w + 1048576);               // 1,048,576 B
  __bf16* whbf    = (__bf16*)(w + 2097152);               // 2,097,152 B
  __bf16* wvt     = (__bf16*)(w + 4194304);               // 33,554,432 B
  __bf16* featbf  = (__bf16*)(w + 37748736);              // 131,072 B
  __bf16* h0buf   = (__bf16*)(w + 37879808);              // 262,144 B
  __bf16* xbf     = (__bf16*)(w + 38141952);              // 4,194,304 B
  float*  xw      = (float*)(w + 42336256);               // 16,777,216 B
  __bf16* hbuf    = (__bf16*)(w + 59113472);              // 8,388,608 B

  hipLaunchKernelGGL(k_zero_out, dim3(1), dim3(1), 0, stream, out);

  hipLaunchKernelGGL(k_f32_to_bf16, dim3(2048), dim3(256), 0, stream, W_proj, wprojbf, DW * HH);
  hipLaunchKernelGGL(k_f32_to_bf16, dim3(2048), dim3(256), 0, stream, Wx, wxbf, DW * HH);
  hipLaunchKernelGGL(k_f32_to_bf16, dim3(4096), dim3(256), 0, stream, Wh, whbf, HH * HH);
  hipLaunchKernelGGL(k_f32_to_bf16, dim3(256),  dim3(256), 0, stream, features, featbf, NB * DW);
  hipLaunchKernelGGL(k_transpose_convert, dim3(VV / 32, HH / 32), dim3(32, 32), 0, stream,
                     W_vocab, wvt);
  hipLaunchKernelGGL(k_embed_gather, dim3((NT * DW) / 256), dim3(256), 0, stream,
                     W_embed, captions, xbf);

  // h0 = features @ W_proj + b_proj        (128x512 @ 512x1024 -> bf16)
  hipLaunchKernelGGL(k_gemm64, dim3(NB / 64, HH / 64), dim3(256), 0, stream,
                     featbf, DW, wprojbf, HH, b_proj, (const float*)nullptr, 0,
                     (float*)nullptr, h0buf, HH, DW, 0);

  // xW = x @ Wx + b                        (4096x512 @ 512x1024 -> f32)
  hipLaunchKernelGGL(k_gemm64, dim3(NT / 64, HH / 64), dim3(256), 0, stream,
                     xbf, DW, wxbf, HH, bvec, (const float*)nullptr, 0,
                     xw, (__bf16*)nullptr, HH, DW, 0);

  // recurrence: h_t = tanh(xW_t + h_{t-1} @ Wh)   (32 sequential 128x1024x1024)
  for (int t = 0; t < TT; ++t) {
    const __bf16* hprev = (t == 0) ? h0buf : (hbuf + (size_t)(t - 1) * NB * HH);
    hipLaunchKernelGGL(k_gemm64, dim3(NB / 64, HH / 64), dim3(256), 0, stream,
                       hprev, HH, whbf, HH, (const float*)nullptr,
                       xw + (size_t)t * NB * HH, HH,
                       (float*)nullptr, hbuf + (size_t)t * NB * HH, HH, HH, 1);
  }

  // fused vocab projection + log-softmax NLL
  hipLaunchKernelGGL(k_vocab_loss, dim3(NT / 16), dim3(256), 0, stream,
                     hbuf, wvt, b_vocab, captions, out);
}
